// CostVolume_66692252172521
// MI455X (gfx1250) — matrix operand order repaired and checked by reading, not compiled, not run on hardware
//
#include <hip/hip_runtime.h>
#include <hip/hip_bf16.h>
#include <math.h>

typedef __attribute__((ext_vector_type(16))) _Float16 v16h;
typedef __attribute__((ext_vector_type(8)))  _Float16 v8h;
typedef __attribute__((ext_vector_type(8)))  float    v8f;

#define B_     8
#define CIN_   64
#define H_     160
#define W_     576
#define COUT_  32
#define MAXD_  48

static __device__ __forceinline__ v16h cat16(v8h lo, v8h hi) {
    return __builtin_shufflevector(lo, hi, 0,1,2,3,4,5,6,7,8,9,10,11,12,13,14,15);
}

// ---------------------------------------------------------------------------
// Kernel 1: fused 3x3 conv (64->32, pad 1) + BN + ReLU + 1x1 conv (32->32)+bias
// One workgroup = one (img, b, h) and a 32-pixel span of W; 64 threads = 2
// wave32s, each wave owns a 16-pixel N-tile x 32 output channels (2 M-tiles).
// conv1 is decomposed as 9 taps x (K=64 over input channels) so that every
// WMMA operand fragment is a pair of contiguous, aligned ds_load_b128:
//   A: sW1r[tap][o][ci]   (W1 repacked channel-contiguous, rows padded to 72h)
//   B: sXT[ky][x][ci]     (input patch channel-contiguous, rows padded to 72h)
// 18 k-steps of v_wmma_f32_16x16x32_f16 per M-tile, f32 accumulation.
// ---------------------------------------------------------------------------
__global__ __launch_bounds__(64)
void feat_kernel(const float* __restrict__ img0, const float* __restrict__ img1,
                 const float* __restrict__ conv_w,
                 const float* __restrict__ bn_gamma, const float* __restrict__ bn_beta,
                 const float* __restrict__ bn_mean,  const float* __restrict__ bn_var,
                 const float* __restrict__ desc_w,   const float* __restrict__ desc_b,
                 _Float16* __restrict__ feat /* [2][B][32][H][W] f16 */)
{
    const int TW = 32;
    // rows padded to 72 halves (144 B): 16B-aligned, 36-bank stride => 16
    // consecutive rows map to 16 distinct start banks (conflict-free b128).
    __shared__ __align__(16) _Float16 sW1r[9 * 32 * 72];   // 41,472 B
    __shared__ __align__(16) _Float16 sXT [3 * 34 * 72];   // 14,688 B
    __shared__ __align__(16) _Float16 sW2 [32 * 40];       //  2,560 B
    __shared__ __align__(16) _Float16 sY1 [2 * 16 * 40];   //  2,560 B
    __shared__ float sInv[32], sShift[32], sBias[32];

    const int tid = threadIdx.x;
    const int w0  = blockIdx.x * TW;
    const int h   = blockIdx.y;
    const int z   = blockIdx.z;
    const int b   = z & 7;
    const int im  = z >> 3;
    const float* __restrict__ src = im ? img1 : img0;

    // Repack W1: sW1r[(t*32+o)*72 + ci] = conv_w[o*576 + ci*9 + t], t = ky*3+kx
    for (int i = tid; i < 9 * 32 * 64; i += 64) {
        int t   = i >> 11;           // / (32*64)
        int rem = i & 2047;
        int o   = rem >> 6;
        int ci  = rem & 63;
        sW1r[(t * 32 + o) * 72 + ci] = (_Float16)conv_w[o * 576 + ci * 9 + t];
    }
    // W2 (32x32), rows padded to 40 halves (80 B, 16B-mult)
    for (int i = tid; i < 32 * 32; i += 64) {
        int o = i >> 5, k = i & 31;
        sW2[o * 40 + k] = (_Float16)desc_w[i];
    }
    // BN affine + bias
    if (tid < 32) {
        float inv = bn_gamma[tid] / sqrtf(bn_var[tid] + 1e-5f);
        sInv[tid]   = inv;
        sShift[tid] = bn_beta[tid] - bn_mean[tid] * inv;
        sBias[tid]  = desc_b[tid];
    }
    // Input patch, channel-contiguous: sXT[(ky*34+x)*72 + ci]
    // (x innermost in the load index for coalesced global reads)
    for (int i = tid; i < 3 * 64 * 34; i += 64) {
        int row = i / 34;            // ky*64 + ci
        int x   = i - row * 34;
        int ky  = row >> 6;
        int ci  = row & 63;
        int gh  = h + ky - 1;
        int gw  = w0 + x - 1;
        float v = 0.0f;
        if ((unsigned)gh < (unsigned)H_ && (unsigned)gw < (unsigned)W_)
            v = src[(((size_t)b * CIN_ + ci) * H_ + gh) * W_ + gw];
        sXT[(ky * 34 + x) * 72 + ci] = (_Float16)v;
    }
    __syncthreads();

    const int wave = tid >> 5;
    const int lane = tid & 31;
    const int l16  = lane & 15;
    const int hlf  = lane >> 4;          // 0 or 1
    const int px   = wave * 16 + l16;    // pixel within tile (N column)

    v8f acc0 = {0.f,0.f,0.f,0.f,0.f,0.f,0.f,0.f};
    v8f acc1 = {0.f,0.f,0.f,0.f,0.f,0.f,0.f,0.f};

    // ---- conv1: 9 taps x 2 k-steps of 32 channels ----
    for (int t = 0; t < 9; ++t) {
        const int ky = t / 3;
        const int kx = t - ky * 3;
        const _Float16* brow = &sXT[(ky * 34 + px + kx) * 72];
#pragma unroll
        for (int ks = 0; ks < 2; ++ks) {
            // A fragments: per-lane K = {hlf*8..+7} u {16+hlf*8..+7} within chunk
            const _Float16* a0p = &sW1r[(t * 32 +  0 + l16) * 72 + ks * 32 + hlf * 8];
            const _Float16* a1p = &sW1r[(t * 32 + 16 + l16) * 72 + ks * 32 + hlf * 8];
            v16h a0 = cat16(*(const v8h*)(a0p), *(const v8h*)(a0p + 16));
            v16h a1 = cat16(*(const v8h*)(a1p), *(const v8h*)(a1p + 16));
            // B fragment: per-lane K = {hlf*8..+7} u {16+hlf*8..+7} within chunk
            const _Float16* bp = brow + ks * 32 + hlf * 8;
            v16h bf = cat16(*(const v8h*)(bp), *(const v8h*)(bp + 16));

            acc0 = __builtin_amdgcn_wmma_f32_16x16x32_f16(false, a0, false, bf,
                                                          (short)0, acc0, false, false);
            acc1 = __builtin_amdgcn_wmma_f32_16x16x32_f16(false, a1, false, bf,
                                                          (short)0, acc1, false, false);
        }
    }

    // ---- BN + ReLU, restage transposed for conv2's B operand ----
#pragma unroll
    for (int v = 0; v < 8; ++v) {
        int c0 =  0 + v + 8 * hlf;
        int c1 = 16 + v + 8 * hlf;
        float y0 = acc0[v] * sInv[c0] + sShift[c0]; y0 = y0 > 0.f ? y0 : 0.f;
        float y1 = acc1[v] * sInv[c1] + sShift[c1]; y1 = y1 > 0.f ? y1 : 0.f;
        sY1[(wave * 16 + l16) * 40 + c0] = (_Float16)y0;
        sY1[(wave * 16 + l16) * 40 + c1] = (_Float16)y1;
    }
    // Same-wave LDS RAW ordering handled by compiler-inserted s_wait_dscnt.

    // ---- conv2: 1x1, K = 32 (one WMMA per M-tile) ----
    const _Float16* bp2 = &sY1[(wave * 16 + l16) * 40 + hlf * 8];
    v16h b2 = cat16(*(const v8h*)(bp2), *(const v8h*)(bp2 + 16));

    v8f out0 = {0.f,0.f,0.f,0.f,0.f,0.f,0.f,0.f};
    v8f out1 = {0.f,0.f,0.f,0.f,0.f,0.f,0.f,0.f};
    {
        const _Float16* ap = &sW2[(0 + l16) * 40 + hlf * 8];
        v16h a2 = cat16(*(const v8h*)(ap), *(const v8h*)(ap + 16));
        out0 = __builtin_amdgcn_wmma_f32_16x16x32_f16(false, a2, false, b2,
                                                      (short)0, out0, false, false);
    }
    {
        const _Float16* ap = &sW2[(16 + l16) * 40 + hlf * 8];
        v16h a2 = cat16(*(const v8h*)(ap), *(const v8h*)(ap + 16));
        out1 = __builtin_amdgcn_wmma_f32_16x16x32_f16(false, a2, false, b2,
                                                      (short)0, out1, false, false);
    }

    // ---- bias + store f16 features ----
    const int w = w0 + wave * 16 + l16;
#pragma unroll
    for (int v = 0; v < 8; ++v) {
        int c0 =  0 + v + 8 * hlf;
        int c1 = 16 + v + 8 * hlf;
        size_t i0 = ((((size_t)im * B_ + b) * COUT_ + c0) * H_ + h) * W_ + w;
        size_t i1 = ((((size_t)im * B_ + b) * COUT_ + c1) * H_ + h) * W_ + w;
        feat[i0] = (_Float16)(out0[v] + sBias[c0]);
        feat[i1] = (_Float16)(out1[v] + sBias[c1]);
    }
}

// ---------------------------------------------------------------------------
// Kernel 2: correlation cost volume. One workgroup = (b, h, 288-px tile).
// L tile and R tile (+47 left halo) staged in LDS as f16; each output is a
// 32-channel dot in f32, scaled by 1/32. Pure bandwidth (141 MB store).
// ---------------------------------------------------------------------------
__global__ __launch_bounds__(256)
void costvol_kernel(const _Float16* __restrict__ feat, float* __restrict__ out)
{
    const int TW = 288;
    const int RW = 336;   // 288 + 47 halo, padded to 336
    __shared__ _Float16 sL[32 * 288];
    __shared__ _Float16 sR[32 * 336];

    const int tid = threadIdx.x;
    const int w0  = blockIdx.x * TW;
    const int h   = blockIdx.y;
    const int b   = blockIdx.z;

    for (int i = tid; i < 32 * TW; i += 256) {
        int c = i / TW, x = i - c * TW;
        sL[i] = feat[(((size_t)(0 * B_ + b) * COUT_ + c) * H_ + h) * W_ + w0 + x];
    }
    for (int i = tid; i < 32 * RW; i += 256) {
        int c = i / RW, x = i - c * RW;
        int gw = w0 - 47 + x;
        _Float16 v = (_Float16)0.0f;
        if (x < 335 && (unsigned)gw < (unsigned)W_)
            v = feat[(((size_t)(1 * B_ + b) * COUT_ + c) * H_ + h) * W_ + gw];
        sR[i] = v;
    }
    __syncthreads();

    for (int idx = tid; idx < MAXD_ * TW; idx += 256) {
        int d  = idx / TW;
        int xx = idx - d * TW;
        int w  = w0 + xx;
        float r = 0.0f;
        if (w >= d) {
            int rx = xx + 47 - d;
#pragma unroll
            for (int c = 0; c < 32; ++c)
                r += (float)sL[c * TW + xx] * (float)sR[c * RW + rx];
            r *= 0.03125f;   // 1/32 channel mean
        }
        out[(((size_t)b * MAXD_ + d) * H_ + h) * W_ + w] = r;
    }
}

// ---------------------------------------------------------------------------
extern "C" void kernel_launch(void* const* d_in, const int* in_sizes, int n_in,
                              void* d_out, int out_size, void* d_ws, size_t ws_size,
                              hipStream_t stream)
{
    const float* left     = (const float*)d_in[0];
    const float* right    = (const float*)d_in[1];
    const float* conv_w   = (const float*)d_in[2];
    const float* bn_gamma = (const float*)d_in[3];
    const float* bn_beta  = (const float*)d_in[4];
    const float* bn_mean  = (const float*)d_in[5];
    const float* bn_var   = (const float*)d_in[6];
    const float* desc_w   = (const float*)d_in[7];
    const float* desc_b   = (const float*)d_in[8];
    // d_in[9] = maxdisp (structural, fixed at 48)

    _Float16* feat = (_Float16*)d_ws;  // [2][8][32][160][576] f16 = 94.4 MB

    dim3 g1(W_ / 32, H_, B_ * 2);
    feat_kernel<<<g1, dim3(64), 0, stream>>>(left, right, conv_w,
                                             bn_gamma, bn_beta, bn_mean, bn_var,
                                             desc_w, desc_b, feat);

    dim3 g2(W_ / 288, H_, B_);
    costvol_kernel<<<g2, dim3(256), 0, stream>>>(feat, (float*)d_out);
}
